// MultiHeadAttention_51178830299348
// MI455X (gfx1250) — matrix operand order repaired and checked by reading, not compiled, not run on hardware
//
#include <hip/hip_runtime.h>
#include <hip/hip_bf16.h>

typedef __bf16 bf16;
typedef __attribute__((ext_vector_type(16))) __bf16 v16bf;
typedef __attribute__((ext_vector_type(8)))  __bf16 v8bf;
typedef __attribute__((ext_vector_type(8)))  float  v8f;

#define Bc 2
#define Sc 2048
#define Dc 2048
#define Hc 16
#define HDc 128
#define QSCALE 0.08838834764831845f   /* 1/sqrt(128) */

__device__ __forceinline__ v8f wmma_bf16(v16bf a, v16bf b, v8f c) {
  return __builtin_amdgcn_wmma_f32_16x16x32_bf16(false, a, false, b, (short)0, c, false, false);
}

// A-fragment (16x32 bf16) from a row-major row pointer (contiguous K).
// Layout: lane<16 -> K 0..15 ; lane>=16 -> K 16..31, row = lane%16.
__device__ __forceinline__ v16bf load_afrag(const bf16* rowptr, int lane) {
  const int koff = (lane < 16) ? 0 : 16;
  v8bf lo = *(const v8bf*)(rowptr + koff);
  v8bf hi = *(const v8bf*)(rowptr + koff + 8);
  return __builtin_shufflevector(lo, hi, 0, 1, 2, 3, 4, 5, 6, 7,
                                         8, 9, 10, 11, 12, 13, 14, 15);
}

// ---------------------------------------------------------------- convert
__global__ void f32_to_bf16_kernel(const float* __restrict__ src,
                                   bf16* __restrict__ dst, size_t n) {
  size_t i = (size_t)blockIdx.x * blockDim.x + threadIdx.x;
  if (i < n) dst[i] = (bf16)src[i];
}

// Tiled transpose + downconvert: src [R][C] f32 -> dst [C][R] bf16.
// One-time pass over the weights so GEMM B-staging is contiguous.
__global__ __launch_bounds__(256) void transpose_to_bf16_kernel(
    const float* __restrict__ src, bf16* __restrict__ dst, int R, int C) {
  __shared__ float tile[32][33];
  const int c0 = blockIdx.x * 32, r0 = blockIdx.y * 32;
  const int tx = threadIdx.x & 31, ty = threadIdx.x >> 5;   // 32 x 8
#pragma unroll
  for (int i = 0; i < 4; ++i)
    tile[ty + i * 8][tx] = src[(size_t)(r0 + ty + i * 8) * C + c0 + tx];
  __syncthreads();
#pragma unroll
  for (int i = 0; i < 4; ++i)
    dst[(size_t)(c0 + ty + i * 8) * R + r0 + tx] = (bf16)tile[tx][ty + i * 8];
}

// ---------------------------------------------------------------- GEMM
// C[M,N] = A[M,K] x Bt[N,K]^T (both bf16 row-major in K), f32 accum.
// 256 threads = 8 waves. WG tile 128(M) x 256(N), K-step 32, double-buffered
// LDS, one barrier per k-step. Wave (wm,wn) -> 64x64 sub-tile:
// 4 A-frags x 4 B-frags -> 16 WMMA per k-step, 16 ds_load_b128 (1:1).
template <bool OUT_F32>
__global__ __launch_bounds__(256) void gemm_bf16_kernel(
    const bf16* __restrict__ A, const bf16* __restrict__ Bt,
    void* __restrict__ Cout, int M, int N, int K) {
  __shared__ __align__(32) bf16 As[2][128][32];   // [buf][m][k]  16 KB
  __shared__ __align__(32) bf16 Bs[2][256][32];   // [buf][n][k]  32 KB

  const int tid  = threadIdx.x;
  const int wave = tid >> 5;
  const int lane = tid & 31;
  const int wm   = wave >> 2;          // 0..1  (M sub-tile, 64 rows)
  const int wn   = wave & 3;           // 0..3  (N sub-tile, 64 cols)
  const int m0   = blockIdx.y * 128;
  const int n0   = blockIdx.x * 256;

  v8f acc[4][4];
#pragma unroll
  for (int i = 0; i < 4; ++i)
#pragma unroll
    for (int j = 0; j < 4; ++j) acc[i][j] = (v8f){};

  const int ar = tid >> 1;             // A stage: row 0..127, 32 B each
  const int ac = (tid & 1) * 16;
  const int bn = tid;                  // B stage: Bt row 0..255, 64 B each

  auto stage = [&](int k0, int buf) {
    { // A: 32 B contiguous per thread
      const v8bf* src = (const v8bf*)(A + (size_t)(m0 + ar) * K + k0 + ac);
      v8bf* dst = (v8bf*)&As[buf][ar][ac];
      dst[0] = src[0];
      dst[1] = src[1];
    }
    { // B: 64 B contiguous per thread (weights pre-transposed to [N][K])
      const v8bf* src = (const v8bf*)(Bt + (size_t)(n0 + bn) * K + k0);
      v8bf* dst = (v8bf*)&Bs[buf][bn][0];
#pragma unroll
      for (int i = 0; i < 4; ++i) dst[i] = src[i];
    }
    if (k0 + 64 < K) { // CDNA5 prefetch of the tile after next
      __builtin_prefetch(A + (size_t)(m0 + ar) * K + k0 + 64 + ac, 0, 1);
      __builtin_prefetch(Bt + (size_t)(n0 + bn) * K + k0 + 64, 0, 1);
    }
  };

  const int nk = K / 32;
  stage(0, 0);
  __syncthreads();
  for (int kk = 0; kk < nk; ++kk) {
    if (kk + 1 < nk) stage((kk + 1) * 32, (kk + 1) & 1);
    const int buf = kk & 1;
    v16bf af[4];
#pragma unroll
    for (int mf = 0; mf < 4; ++mf)
      af[mf] = load_afrag(&As[buf][wm * 64 + mf * 16 + (lane & 15)][0], lane);
    const int kb = (lane < 16) ? 0 : 16;
#pragma unroll
    for (int nt = 0; nt < 4; ++nt) {
      v16bf bf = *(const v16bf*)&Bs[buf][wn * 64 + nt * 16 + (lane & 15)][kb];
#pragma unroll
      for (int mf = 0; mf < 4; ++mf)
        acc[mf][nt] = wmma_bf16(af[mf], bf, acc[mf][nt]);
    }
    __syncthreads();
  }

  const int half = lane >> 4;
#pragma unroll
  for (int mf = 0; mf < 4; ++mf)
#pragma unroll
    for (int nt = 0; nt < 4; ++nt)
#pragma unroll
      for (int r = 0; r < 8; ++r) {
        const int row = m0 + wm * 64 + mf * 16 + r + half * 8;
        const int col = n0 + wn * 64 + nt * 16 + (lane & 15);
        const float v = acc[mf][nt][r];
        if (OUT_F32) ((float*)Cout)[(size_t)row * N + col] = v;
        else         ((bf16*)Cout)[(size_t)row * N + col] = (bf16)v;
      }
}

// ---------------------------------------------------------------- RoPE + relayout
// qkv [B*S, 3D] bf16 -> Q,K [B*H, S, 128] bf16 (Q pre-scaled), Vt [B*H, 128, S]
__global__ __launch_bounds__(128) void rope_kernel(
    const bf16* __restrict__ qkv, const float* __restrict__ sinp,
    const float* __restrict__ cosp, bf16* __restrict__ Q,
    bf16* __restrict__ K, bf16* __restrict__ Vt) {
  const int d  = threadIdx.x;            // 0..127
  const int bh = blockIdx.x / Sc;        // b*H + h
  const int s  = blockIdx.x % Sc;
  const int b  = bh / Hc;
  const int h  = bh % Hc;

  const size_t rowbase = ((size_t)b * Sc + s) * (size_t)(3 * Dc);
  const int col = h * HDc + d;
  const float q  = (float)qkv[rowbase + col];
  const float k  = (float)qkv[rowbase + Dc + col];
  const float v  = (float)qkv[rowbase + 2 * Dc + col];
  const int   dp = (d < 64) ? (d + 64) : (d - 64);
  const float qp = (float)qkv[rowbase + h * HDc + dp];
  const float kp = (float)qkv[rowbase + Dc + h * HDc + dp];
  const float sg = (d < 64) ? -1.0f : 1.0f;
  const float sn = sinp[s * HDc + d];
  const float cs = cosp[s * HDc + d];

  const size_t o = ((size_t)bh * Sc + s) * HDc + d;
  Q[o] = (bf16)((q * cs + sg * qp * sn) * QSCALE);
  K[o] = (bf16)(k * cs + sg * kp * sn);
  Vt[((size_t)bh * HDc + d) * Sc + s] = (bf16)v;
}

// ---------------------------------------------------------------- flash attention
// WG: 128 thr = 4 waves; WG owns (b,h) and 64 q rows; wave w -> 16 q rows.
// KV processed in 32-key blocks shared via LDS; causal.
__global__ __launch_bounds__(128) void attn_kernel(
    const bf16* __restrict__ Q, const bf16* __restrict__ Kt,
    const bf16* __restrict__ Vt, bf16* __restrict__ O) {
  __shared__ __align__(32) bf16 Ks[32][128];     // [kv][d]   8 KB
  __shared__ __align__(32) bf16 Vs[128][32];     // [d][kv]   8 KB
  __shared__ __align__(32) bf16 Ps[4][16][32];   // per-wave P transpose, 4 KB

  const int tid  = threadIdx.x;
  const int wave = tid >> 5;
  const int lane = tid & 31;
  const int half = lane >> 4;
  const int bh   = blockIdx.x;           // b*H + h
  const int b    = bh / Hc;
  const int h    = bh % Hc;
  const int q0   = blockIdx.y * 64;
  const int qw   = q0 + wave * 16;

  const bf16* Qb = Q  + (size_t)bh * Sc * HDc;
  const bf16* Kb = Kt + (size_t)bh * Sc * HDc;
  const bf16* Vb = Vt + (size_t)bh * HDc * Sc;

  // persistent Q fragments over HD=128 (4 x 16x32)
  v16bf qf[4];
  {
    const bf16* rp = Qb + (size_t)(qw + (lane & 15)) * HDc;
#pragma unroll
    for (int f = 0; f < 4; ++f) qf[f] = load_afrag(rp + f * 32, lane);
  }

  v8f acc[8];
#pragma unroll
  for (int i = 0; i < 8; ++i) acc[i] = (v8f){};
  float m_i[8], l_i[8];
#pragma unroll
  for (int r = 0; r < 8; ++r) { m_i[r] = -1e30f; l_i[r] = 0.0f; }

  const int nkb = (q0 + 64) / 32;        // causal: keys up to WG's last q row
  for (int kb = 0; kb < nkb; ++kb) {
    const int kv0 = kb * 32;
    { // stage K block (32x128): 64 B contiguous per thread
      const int r = tid >> 2, c = (tid & 3) * 32;
      const v8bf* src = (const v8bf*)(Kb + (size_t)(kv0 + r) * HDc + c);
      v8bf* dst = (v8bf*)&Ks[r][c];
#pragma unroll
      for (int i = 0; i < 4; ++i) dst[i] = src[i];
    }
    { // stage V^T block (128x32): 64 B contiguous per thread
      const v8bf* src = (const v8bf*)(Vb + (size_t)tid * Sc + kv0);
      v8bf* dst = (v8bf*)&Vs[tid][0];
#pragma unroll
      for (int i = 0; i < 4; ++i) dst[i] = src[i];
    }
    __syncthreads();

    // S = Q K^T : two 16x16 tiles, 8 WMMAs
    v8f s0 = (v8f){}, s1 = (v8f){};
    const int ksb = half * 16;
#pragma unroll
    for (int f = 0; f < 4; ++f) {
      v16bf b0 = *(const v16bf*)&Ks[(lane & 15)][f * 32 + ksb];
      v16bf b1 = *(const v16bf*)&Ks[16 + (lane & 15)][f * 32 + ksb];
      s0 = wmma_bf16(qf[f], b0, s0);
      s1 = wmma_bf16(qf[f], b1, s1);
    }

    // causal mask + online softmax (row stats live in 16-lane halves)
    float p0[8], p1[8];
#pragma unroll
    for (int r = 0; r < 8; ++r) {
      const int qrow = qw + r + half * 8;
      const int c0 = kv0 + (lane & 15);
      const float v0 = (c0      <= qrow) ? s0[r] : -1e30f;
      const float v1 = (c0 + 16 <= qrow) ? s1[r] : -1e30f;
      float mx = fmaxf(v0, v1);
#pragma unroll
      for (int o = 8; o >= 1; o >>= 1) mx = fmaxf(mx, __shfl_xor(mx, o));
      const float mnew = fmaxf(m_i[r], mx);
      const float sc   = __expf(m_i[r] - mnew);
      const float e0   = __expf(v0 - mnew);
      const float e1   = __expf(v1 - mnew);
      float rs = e0 + e1;
#pragma unroll
      for (int o = 8; o >= 1; o >>= 1) rs += __shfl_xor(rs, o);
      l_i[r] = l_i[r] * sc + rs;
      m_i[r] = mnew;
      p0[r] = e0; p1[r] = e1;
#pragma unroll
      for (int nt = 0; nt < 8; ++nt) acc[nt][r] *= sc;
    }

    // transpose P from C/D layout to A layout via per-wave LDS
#pragma unroll
    for (int r = 0; r < 8; ++r) {
      Ps[wave][r + half * 8][(lane & 15)]      = (bf16)p0[r];
      Ps[wave][r + half * 8][16 + (lane & 15)] = (bf16)p1[r];
    }
    asm volatile("s_wait_dscnt 0" ::: "memory");
    v16bf pf = load_afrag(&Ps[wave][(lane & 15)][0], lane);

    // O += P V : 8 WMMAs over HD=128
#pragma unroll
    for (int nt = 0; nt < 8; ++nt) {
      v16bf bv = *(const v16bf*)&Vs[nt * 16 + (lane & 15)][half * 16];
      acc[nt] = wmma_bf16(pf, bv, acc[nt]);
    }
    __syncthreads();
  }

  // normalize + write O in [B,S,D] so the projection GEMM reads row-major
  bf16* Ob = O + (size_t)b * Sc * Dc;
#pragma unroll
  for (int r = 0; r < 8; ++r) {
    const int qrow = qw + r + half * 8;
    const float inv = 1.0f / l_i[r];
#pragma unroll
    for (int nt = 0; nt < 8; ++nt) {
      const int d = nt * 16 + (lane & 15);
      Ob[(size_t)qrow * Dc + h * HDc + d] = (bf16)(acc[nt][r] * inv);
    }
  }
}

// ---------------------------------------------------------------- launch
extern "C" void kernel_launch(void* const* d_in, const int* in_sizes, int n_in,
                              void* d_out, int out_size, void* d_ws, size_t ws_size,
                              hipStream_t stream) {
  const float* x     = (const float*)d_in[0];
  /* d_in[1] = mask (causal tril, handled analytically) */
  const float* sinp  = (const float*)d_in[2];
  const float* cosp  = (const float*)d_in[3];
  const float* Wqkv  = (const float*)d_in[4];
  const float* Wproj = (const float*)d_in[5];
  float* out = (float*)d_out;

  const size_t nx    = (size_t)Bc * Sc * Dc;        // 8388608
  const size_t nwq   = (size_t)Dc * 3 * Dc;         // 12582912
  const size_t nwp   = (size_t)Dc * Dc;             // 4194304
  const size_t nqkv  = (size_t)Bc * Sc * 3 * Dc;    // 25165824
  const size_t nhead = (size_t)Bc * Hc * Sc * HDc;  // 8388608

  char* ws = (char*)d_ws;
  bf16* xb   = (bf16*)ws; ws += nx    * 2;
  bf16* wqbT = (bf16*)ws; ws += nwq   * 2;          // Wqkv^T  [3D][D]
  bf16* wpbT = (bf16*)ws; ws += nwp   * 2;          // Wproj^T [D][D]
  bf16* qkvb = (bf16*)ws; ws += nqkv  * 2;
  bf16* Qr   = (bf16*)ws; ws += nhead * 2;
  bf16* Kr   = (bf16*)ws; ws += nhead * 2;
  bf16* Vt   = (bf16*)ws; ws += nhead * 2;
  bf16* Ob   = (bf16*)ws; ws += nx    * 2;          // ~168 MB total

  // 1) x -> bf16 ; weights -> bf16 transposed [N][K]
  f32_to_bf16_kernel<<<dim3((nx + 255) / 256), 256, 0, stream>>>(x, xb, nx);
  transpose_to_bf16_kernel<<<dim3(3 * Dc / 32, Dc / 32), 256, 0, stream>>>(
      Wqkv, wqbT, Dc, 3 * Dc);
  transpose_to_bf16_kernel<<<dim3(Dc / 32, Dc / 32), 256, 0, stream>>>(
      Wproj, wpbT, Dc, Dc);

  // 2) qkv = x @ Wqkv    (M=4096, N=6144, K=2048)
  gemm_bf16_kernel<false><<<dim3(3 * Dc / 256, Bc * Sc / 128), 256, 0, stream>>>(
      xb, wqbT, qkvb, Bc * Sc, 3 * Dc, Dc);

  // 3) RoPE + head relayout
  rope_kernel<<<dim3(Bc * Hc * Sc), 128, 0, stream>>>(qkvb, sinp, cosp, Qr, Kr, Vt);

  // 4) causal flash attention
  attn_kernel<<<dim3(Bc * Hc, Sc / 64), 128, 0, stream>>>(Qr, Kr, Vt, Ob);

  // 5) out = O @ Wproj   (M=4096, N=2048, K=2048), f32 output
  gemm_bf16_kernel<true><<<dim3(Dc / 256, Bc * Sc / 128), 256, 0, stream>>>(
      Ob, wpbT, out, Bc * Sc, Dc, Dc);
}
